// EMA_55868934586523
// MI455X (gfx1250) — matrix operand order repaired and checked by reading, not compiled, not run on hardware
//
#include <hip/hip_runtime.h>
#include <hip/hip_bf16.h>

typedef __attribute__((ext_vector_type(16))) _Float16 v16h;
typedef __attribute__((ext_vector_type(8)))  float    v8f;

typedef __attribute__((address_space(1))) int g_int;   // global
typedef __attribute__((address_space(3))) int l_int;   // LDS

#define BG    256        // b*GROUPS
#define CG    16
#define NN    16384
#define TSZ   256        // time-tile per workgroup in apply kernel
#define HSTR  24         // halves per LDS column (16 used, 48B stride keeps 16B align)
#define FSTR  17         // f32 tile column stride (bank-conflict-free)

#if defined(__gfx1250__) && __has_builtin(__builtin_amdgcn_global_load_async_to_lds_b32)
#define EMA_ASYNC_LDS 1
#else
#define EMA_ASYNC_LDS 0
#endif

// ---------------- Pass 1: per (bg, c) sums: S, Q=Σx², first, last -------------
__global__ void ema_stats_kernel(const float* __restrict__ x, float* __restrict__ stats) {
    const int bg  = blockIdx.x;
    const int tid = threadIdx.x;           // 256 threads = 16 channels × 16 strips
    const int c   = tid >> 4;
    const int j   = tid & 15;
    const float* row = x + ((size_t)bg * CG + c) * NN;
    float s = 0.f, q = 0.f;
    for (int it = 0; it < NN / 16; ++it) {
        if ((it & 63) == 0)
            __builtin_prefetch(row + it * 16 + j + 4096, 0, 0);  // global_prefetch_b8
        float v = row[it * 16 + j];
        s += v; q += v * v;
    }
    // reduce across the 16 strip-lanes of each channel (stays inside 16-lane groups)
    for (int o = 1; o < 16; o <<= 1) {
        s += __shfl_xor(s, o, 16);
        q += __shfl_xor(q, o, 16);
    }
    float* st = stats + bg * 64;
    if (j == 0) { st[c] = s; st[16 + c] = q; }
    if (tid < 16)       st[32 + tid]        = x[((size_t)bg * CG + tid) * NN];            // first
    else if (tid < 32)  st[48 + (tid - 16)] = x[((size_t)bg * CG + (tid - 16)) * NN + (NN - 1)]; // last
}

// ---------------- Pass 2: collapse everything into 49 scalars per bg ----------
__global__ void ema_coeff_kernel(const float* __restrict__ stats,
                                 const float* __restrict__ w1, const float* __restrict__ b1,
                                 const float* __restrict__ w3, const float* __restrict__ b3,
                                 const float* __restrict__ gamma, const float* __restrict__ beta,
                                 float* __restrict__ coeff) {
    const int bg = blockIdx.x;
    const int lane = threadIdx.x;          // 32 threads, lanes 0..15 active for channel work
    __shared__ float sMean[CG], sS[CG], sF[CG], sL[CG], sX11[CG];

    const float* st = stats + bg * 64;
    float S = 0.f, Q = 0.f, F = 0.f, L = 0.f;
    if (lane < CG) {
        S = st[lane]; Q = st[16 + lane]; F = st[32 + lane]; L = st[48 + lane];
        sS[lane] = S; sF[lane] = F; sL[lane] = L;
    }
    const float invN = 1.f / (float)NN;
    float mean = S * invN;
    float var  = fmaxf(Q * invN - mean * mean, 0.f);
    if (lane < CG) sMean[lane] = mean;
    __syncthreads();

    float sg = 0.f, rs = 0.f, a = 0.f, x21 = 0.f, x11 = 0.f;
    if (lane < CG) {
        const int c = lane;
        // gate: sigmoid(w1 @ mean + b1)
        float z = b1[c];
        for (int k = 0; k < CG; ++k) z += w1[c * CG + k] * sMean[k];
        sg = 1.f / (1.f + expf(-z));
        rs = rsqrtf(sg * sg * var + 1e-5f);   // rsqrt(var(x1_pre)+eps)
        // mean over n of conv3x3 output (pad-1 boundary corrected)
        float acc = 0.f;
        for (int i = 0; i < CG; ++i) {
            const float* w = w3 + (c * CG + i) * 3;
            acc += w[0] * (sS[i] - sL[i]) + w[1] * sS[i] + w[2] * (sS[i] - sF[i]);
        }
        float x2m = b3[c] + acc * invN;
        // softmax over channels of x2-mean  -> x21
        float m = x2m;
        for (int o = 8; o > 0; o >>= 1) m = fmaxf(m, __shfl_xor(m, o, 16));
        float e = expf(x2m - m); float se = e;
        for (int o = 8; o > 0; o >>= 1) se += __shfl_xor(se, o, 16);
        x21 = e / se;
        // mean over n of x1 is exactly beta  -> x11 = softmax(beta)
        float bb = beta[c];
        float mb = bb;
        for (int o = 8; o > 0; o >>= 1) mb = fmaxf(mb, __shfl_xor(mb, o, 16));
        float eb = expf(bb - mb); float seb = eb;
        for (int o = 8; o > 0; o >>= 1) seb += __shfl_xor(seb, o, 16);
        x11 = eb / seb;
        sX11[c] = x11;
        a = x21 * gamma[c] * rs * sg;         // x1-branch coefficient on g[c,t]
    }
    __syncthreads();
    if (lane < CG) {
        const int i = lane;
        // v[i,k] = sum_c x11[c] * w3[c,i,k]
        float v0 = 0.f, v1 = 0.f, v2 = 0.f;
        for (int c2 = 0; c2 < CG; ++c2) {
            float w11 = sX11[c2];
            const float* w = w3 + (c2 * CG + i) * 3;
            v0 += w11 * w[0]; v1 += w11 * w[1]; v2 += w11 * w[2];
        }
        float* co = coeff + bg * 64;
        co[i]      = v0;         // tap on g[i, t-1]
        co[16 + i] = v1 + a;     // tap on g[i, t]   (conv center + x1 branch)
        co[32 + i] = v2;         // tap on g[i, t+1]
        // constant: x11·b3 + Σ x21·(beta - γ·rs·s·mean)
        float t = x11 * b3[lane] + x21 * (beta[lane] - gamma[lane] * rs * sg * mean);
        float Ct = t;
        for (int o = 8; o > 0; o >>= 1) Ct += __shfl_xor(Ct, o, 16);
        if (lane == 0) co[48] = Ct;
    }
}

// ---------------- Pass 3: async fill, stencil-reduce via WMMA, gate, store ----
__global__ void ema_apply_kernel(const float* __restrict__ x,
                                 const float* __restrict__ coeff,
                                 float* __restrict__ out) {
    __shared__ __align__(64) _Float16 tileH[(TSZ + 2) * HSTR]; // col-major f16 + halo
    __shared__ __align__(64) float tileF[TSZ * FSTR];          // col-major f32 (exact copy)

    const int bg   = blockIdx.y;
    const int tb   = blockIdx.x * TSZ;
    const int tid  = threadIdx.x;            // 256 threads = 8 waves
    const int lane = tid & 31;
    const size_t base = (size_t)bg * CG * NN;

#if EMA_ASYNC_LDS
    // Fill f32 tile with per-lane async global->LDS scatter (no VGPR round-trip).
    // Global side: consecutive lanes -> consecutive addresses (coalesced).
    // LDS side: stride-17 column-major scatter (bank-conflict-free).
    for (int idx = tid; idx < CG * TSZ; idx += 256) {
        int c  = idx >> 8;                   // idx / TSZ
        int lc = idx & (TSZ - 1);
        const float* gp = x + base + (size_t)c * NN + tb + lc;
        __builtin_amdgcn_global_load_async_to_lds_b32(
            (g_int*)gp, (l_int*)&tileF[lc * FSTR + c], 0, 0);
    }
#else
    for (int idx = tid; idx < CG * TSZ; idx += 256) {
        int c  = idx >> 8;
        int lc = idx & (TSZ - 1);
        tileF[lc * FSTR + c] = x[base + (size_t)c * NN + tb + lc];
    }
#endif
    // halo columns (zero pad at sequence ends) -- f16 tile only
    if (tid < 16) {
        float v = (tb > 0) ? x[base + (size_t)tid * NN + tb - 1] : 0.f;
        tileH[tid] = (_Float16)v;
    } else if (tid < 32) {
        int c = tid - 16;
        float v = (tb + TSZ < NN) ? x[base + (size_t)c * NN + tb + TSZ] : 0.f;
        tileH[(TSZ + 1) * HSTR + c] = (_Float16)v;
    }

    // Build A operands (identical rows => per-lane content depends only on lane half).
    // WMMA#1: K=0..15 -> (i=K, tap-1) coeff u0 ; K=16..31 -> (i=K-16, tap0) coeff u1
    // WMMA#2: K=0..15 -> (i=K, tap+1) coeff u2 ; K=16..31 -> zero
    const float* cf = coeff + bg * 64;
    const int hb = (lane < 16) ? 0 : 16;
    v16h A1, A2;
#pragma unroll
    for (int j = 0; j < 16; ++j) {
        A1[j]     = (_Float16)cf[hb + j];        // u0[0..15] (lo half) / u1[0..15] (hi half)
        A2[j]     = (lane < 16) ? (_Float16)cf[32 + j] : (_Float16)0.f;   // u2[0..15] / zero
    }
    const float Cbg = cf[48];

#if EMA_ASYNC_LDS
#if __has_builtin(__builtin_amdgcn_s_wait_asynccnt)
    __builtin_amdgcn_s_wait_asynccnt(0);
#else
    asm volatile("s_wait_asynccnt 0x0" ::: "memory");
#endif
#endif
    __syncthreads();
    // Convert the landed f32 tile into the col-major f16 tile (LDS->LDS).
    for (int idx = tid; idx < CG * TSZ; idx += 256) {
        int c  = idx >> 8;
        int lc = idx & (TSZ - 1);
        tileH[(lc + 1) * HSTR + c] = (_Float16)tileF[lc * FSTR + c];
    }
    __syncthreads();

    const int w = tid >> 5;
#pragma unroll
    for (int sI = 0; sI < 2; ++sI) {
        const int t0 = (w * 2 + sI) * 16;        // 16-column slab of this wave
        // B layout (32x16 f16): lanes 0-15 carry K=0..15 of column N=lane,
        // lanes 16-31 carry K=16..31 of column N=lane-16.
        const int hc1 = (lane < 16) ? (t0 + lane)            // g[:, t-1] for K=0..15
                                    : (t0 + (lane - 16) + 1);// g[:, t]   for K=16..31
        const int hc2 = (lane < 16) ? (t0 + lane + 2)        // g[:, t+1]
                                    : (t0 + (lane - 16) + 1);// don't care: A is zero there
        v16h B1 = *(const v16h*)(tileH + hc1 * HSTR);
        v16h B2 = *(const v16h*)(tileH + hc2 * HSTR);
        v8f acc = {};
        acc = __builtin_amdgcn_wmma_f32_16x16x32_f16(false, A1, false, B1,
                                                     (short)0, acc, false, false);
        acc = __builtin_amdgcn_wmma_f32_16x16x32_f16(false, A2, false, B2,
                                                     (short)0, acc, false, false);
        // every D row is identical; VGPR0 holds this lane's t = t0 + (lane&15)
        const float wt  = acc[0] + Cbg;
        const float gte = 1.f / (1.f + expf(-wt));
        const int   lt  = t0 + (lane & 15);
        const int   cb  = (lane < 16) ? 0 : 8;   // split channels so halves don't duplicate stores
#pragma unroll
        for (int cc = 0; cc < 8; ++cc) {
            int c = cb + cc;
            out[base + (size_t)c * NN + tb + lt] = tileF[lt * FSTR + c] * gte;
        }
    }
}

extern "C" void kernel_launch(void* const* d_in, const int* in_sizes, int n_in,
                              void* d_out, int out_size, void* d_ws, size_t ws_size,
                              hipStream_t stream) {
    const float* x     = (const float*)d_in[0];
    const float* w1    = (const float*)d_in[1];
    const float* b1    = (const float*)d_in[2];
    const float* w3    = (const float*)d_in[3];
    const float* b3    = (const float*)d_in[4];
    const float* gamma = (const float*)d_in[5];
    const float* beta  = (const float*)d_in[6];
    float* out   = (float*)d_out;
    float* stats = (float*)d_ws;               // BG*64 floats
    float* coeff = stats + BG * 64;            // BG*64 floats

    ema_stats_kernel<<<BG, 256, 0, stream>>>(x, stats);
    ema_coeff_kernel<<<BG, 32, 0, stream>>>(stats, w1, b1, w3, b3, gamma, beta, coeff);
    ema_apply_kernel<<<dim3(NN / TSZ, BG), 256, 0, stream>>>(x, coeff, out);
}
